// LSTMRNN_81578608820397
// MI455X (gfx1250) — matrix operand order repaired and checked by reading, not compiled, not run on hardware
//
#include <hip/hip_runtime.h>
#include <math.h>

// ---------------------------------------------------------------------------
// Seq2seq LSTM (2-layer enc + 2-layer greedy dec), HID=1024, VOCAB=32000,
// SEQ=64, batch=1.  Strategy: bandwidth-bound GEMV chain -> convert weights
// to bf16 once (halves traffic; ~126MB total fits MI455X's 192MB L2), then
// drive every matvec through v_wmma_f32_16x16x32_bf16 with weights
// pre-swizzled into a contiguous-k (k = 16*half + j) A-fragment order.
// ---------------------------------------------------------------------------

typedef __attribute__((ext_vector_type(16))) __bf16 v16bf;
typedef __attribute__((ext_vector_type(8)))  float  v8f;

#define HID    1024
#define GATES  4096
#define VOCAB  32000
#define SEQ    64
#define NL     2
#define KB     32          // K chunks of 32 per 1024-deep dot product
#define TILE_ELEMS 512     // 16 rows x 32 k = one A fragment

__device__ __forceinline__ unsigned short f32_to_bf16(float f) {
    unsigned int u = __float_as_uint(f);
    u += 0x7fffu + ((u >> 16) & 1u);           // round-to-nearest-even
    return (unsigned short)(u >> 16);
}

// ---------------------------------------------------------------------------
// One-time weight swizzle: row-major f32 [R,1024] -> bf16 tiles in the exact
// per-lane order this file uses for the 16-bit A 16x32 fragment (k contiguous per half):
//   lanes 0-15  hold rows m=lane,    VGPR0..7 = K0..15
//   lanes 16-31 hold rows m=lane-16, VGPR0..7 = K16..31
// Tile (rb,kb) stored at ((rb*32)+kb)*512; element e = lane*16 + j.
// ---------------------------------------------------------------------------
__global__ void convert_tiled_bf16(const float* __restrict__ W,
                                   unsigned short* __restrict__ Wt, int R) {
    long e = (long)blockIdx.x * blockDim.x + threadIdx.x;
    long total = (long)R * HID;
    if (e >= total) return;
    int  within = (int)(e & (TILE_ELEMS - 1));
    long tile   = e >> 9;
    int  l = within >> 4;          // lane 0..31
    int  j = within & 15;          // element within lane
    long rb = tile >> 5;
    int  kb = (int)(tile & 31);
    int  m = (int)(rb * 16) + (l & 15);
    int  k = kb * 32 + ((l >> 4) << 4) + j;
    Wt[e] = f32_to_bf16(W[(long)m * HID + k]);
}

// ---------------------------------------------------------------------------
// WMMA GEMV: y[rb*16 .. rb*16+15] = W1*x1 (+ W2*x2) (+ b0) (+ b1)
// One wave per 16-row block; B fragment = x chunk replicated over 16 columns
// (lanes 0-15 broadcast x[k0..k0+15], lanes 16-31 x[k0+16..k0+31], 2 halves
// per VGPR -> same contiguous k order as the swizzled A tiles).  Lanes 0/16 (column 0)
// extract rows 0-7 / 8-15 from the f32 C/D fragment.
// ---------------------------------------------------------------------------
__global__ void wmma_gemv(const unsigned short* __restrict__ W1t,
                          const unsigned short* __restrict__ x1,
                          const unsigned short* __restrict__ W2t,
                          const unsigned short* __restrict__ x2,
                          const float* __restrict__ b0,
                          const float* __restrict__ b1,
                          float* __restrict__ y, int nRB) {
    int wave = blockIdx.x * (blockDim.x >> 5) + (threadIdx.x >> 5);
    int lane = threadIdx.x & 31;
    if (wave >= nRB) return;   // grids sized exactly; never taken (EXEC stays all-1s)

    const __bf16* A1 = reinterpret_cast<const __bf16*>(W1t)
                       + (long)wave * (KB * TILE_ELEMS) + lane * 16;
    const __bf16* X1 = reinterpret_cast<const __bf16*>(x1) + ((lane >> 4) << 4);
    v8f acc = {};

    if (W2t) {
        const __bf16* A2 = reinterpret_cast<const __bf16*>(W2t)
                           + (long)wave * (KB * TILE_ELEMS) + lane * 16;
        const __bf16* X2 = reinterpret_cast<const __bf16*>(x2) + ((lane >> 4) << 4);
        #pragma unroll 4
        for (int kb = 0; kb < KB; ++kb) {
            v16bf a1 = *(const v16bf*)(A1 + kb * TILE_ELEMS);
            v16bf v1 = *(const v16bf*)(X1 + kb * 32);
            acc = __builtin_amdgcn_wmma_f32_16x16x32_bf16(
                      false, a1, false, v1, (short)0, acc, false, false);
            v16bf a2 = *(const v16bf*)(A2 + kb * TILE_ELEMS);
            v16bf v2 = *(const v16bf*)(X2 + kb * 32);
            acc = __builtin_amdgcn_wmma_f32_16x16x32_bf16(
                      false, a2, false, v2, (short)0, acc, false, false);
        }
    } else {
        #pragma unroll 4
        for (int kb = 0; kb < KB; ++kb) {
            v16bf a1 = *(const v16bf*)(A1 + kb * TILE_ELEMS);
            v16bf v1 = *(const v16bf*)(X1 + kb * 32);
            acc = __builtin_amdgcn_wmma_f32_16x16x32_bf16(
                      false, a1, false, v1, (short)0, acc, false, false);
        }
    }

    if ((lane & 15) == 0) {
        int mbase = wave * 16 + ((lane >> 4) << 3);
        #pragma unroll
        for (int r = 0; r < 8; ++r) {
            float v = acc[r];
            if (b0) v += b0[mbase + r];
            if (b1) v += b1[mbase + r];
            y[mbase + r] = v;
        }
    }
}

// ---------------------------------------------------------------------------
// Pointwise LSTM cell: g=[i|f|g|o] (4096 f32) -> updates c (f32), h (bf16).
// ---------------------------------------------------------------------------
__global__ void lstm_cell(const float* __restrict__ g,
                          float* __restrict__ c,
                          unsigned short* __restrict__ h) {
    int j = blockIdx.x * blockDim.x + threadIdx.x;
    if (j >= HID) return;
    float gi = g[j], gf = g[HID + j], gg = g[2 * HID + j], go = g[3 * HID + j];
    float i = 1.f / (1.f + expf(-gi));
    float f = 1.f / (1.f + expf(-gf));
    float cand = tanhf(gg);
    float o = 1.f / (1.f + expf(-go));
    float cc = f * c[j] + i * cand;
    c[j] = cc;
    h[j] = f32_to_bf16(o * tanhf(cc));
}

// Embedding row gather -> bf16 input vector.
__global__ void embed_bf16(const float* __restrict__ emb,
                           const int* __restrict__ tokp,
                           unsigned short* __restrict__ x) {
    int j = blockIdx.x * blockDim.x + threadIdx.x;
    if (j >= HID) return;
    int tok = *tokp;
    x[j] = f32_to_bf16(emb[(long)tok * HID + j]);
}

// Zero h/c state, seed decoder token slot with BOS.
__global__ void init_state(unsigned short* __restrict__ h,
                           float* __restrict__ c,
                           int* __restrict__ tok) {
    int j = blockIdx.x * blockDim.x + threadIdx.x;
    if (j < NL * HID) { h[j] = 0; c[j] = 0.f; }
    if (j == 0) *tok = 1;  // BOS
}

// log_softmax over 32000 logits + greedy argmax (first-max tie-break).
__global__ void logsoftmax_argmax(const float* __restrict__ logits,
                                  float* __restrict__ outrow,
                                  int* __restrict__ tok) {
    __shared__ float sv[256];
    __shared__ int   si[256];
    int tid = threadIdx.x;

    float best = -3.402823466e38f; int bi = 0;
    for (int v = tid; v < VOCAB; v += 256) {
        float x = logits[v];
        if (x > best) { best = x; bi = v; }
    }
    sv[tid] = best; si[tid] = bi;
    __syncthreads();
    for (int s = 128; s > 0; s >>= 1) {
        if (tid < s) {
            if (sv[tid + s] > sv[tid] ||
                (sv[tid + s] == sv[tid] && si[tid + s] < si[tid])) {
                sv[tid] = sv[tid + s]; si[tid] = si[tid + s];
            }
        }
        __syncthreads();
    }
    float mx = sv[0];
    int amax = si[0];
    __syncthreads();

    float sum = 0.f;
    for (int v = tid; v < VOCAB; v += 256) sum += expf(logits[v] - mx);
    sv[tid] = sum;
    __syncthreads();
    for (int s = 128; s > 0; s >>= 1) {
        if (tid < s) sv[tid] += sv[tid + s];
        __syncthreads();
    }
    float lz = mx + logf(sv[0]);

    for (int v = tid; v < VOCAB; v += 256) outrow[v] = logits[v] - lz;
    if (tid == 0) *tok = amax;
}

// ---------------------------------------------------------------------------
// Workspace layout (bytes, all 256-aligned).  Total ~132.8 MB.
// ---------------------------------------------------------------------------
#define WB_ENC (8192L * 1024L * 2L)          // 16 MB per [2,4096,1024] bf16
static const size_t OFF_ENC_WIH = 0;
static const size_t OFF_ENC_WHH = OFF_ENC_WIH + WB_ENC;
static const size_t OFF_DEC_WIH = OFF_ENC_WHH + WB_ENC;
static const size_t OFF_DEC_WHH = OFF_DEC_WIH + WB_ENC;
static const size_t OFF_OUTW    = OFF_DEC_WHH + WB_ENC;          // 65,536,000 B
static const size_t OFF_X       = OFF_OUTW + 32000L * 1024L * 2L;
static const size_t OFF_H       = OFF_X + 2048;
static const size_t OFF_C       = OFF_H + 4096;
static const size_t OFF_G       = OFF_C + 8192;
static const size_t OFF_LOGITS  = OFF_G + 16384;
static const size_t OFF_TOK     = OFF_LOGITS + 128000;

#define LAYER_WT_ELEMS (4096L * 1024L)       // bf16 elems per layer matrix

extern "C" void kernel_launch(void* const* d_in, const int* in_sizes, int n_in,
                              void* d_out, int out_size, void* d_ws, size_t ws_size,
                              hipStream_t stream) {
    const int*   src     = (const int*)  d_in[0];
    const float* emb_enc = (const float*)d_in[2];
    const float* enc_Wih = (const float*)d_in[3];
    const float* enc_Whh = (const float*)d_in[4];
    const float* enc_bih = (const float*)d_in[5];
    const float* enc_bhh = (const float*)d_in[6];
    const float* emb_dec = (const float*)d_in[7];
    const float* dec_Wih = (const float*)d_in[8];
    const float* dec_Whh = (const float*)d_in[9];
    const float* dec_bih = (const float*)d_in[10];
    const float* dec_bhh = (const float*)d_in[11];
    const float* out_W   = (const float*)d_in[12];
    const float* out_b   = (const float*)d_in[13];
    float* out = (float*)d_out;

    char* ws = (char*)d_ws;
    unsigned short* encWih_t = (unsigned short*)(ws + OFF_ENC_WIH);
    unsigned short* encWhh_t = (unsigned short*)(ws + OFF_ENC_WHH);
    unsigned short* decWih_t = (unsigned short*)(ws + OFF_DEC_WIH);
    unsigned short* decWhh_t = (unsigned short*)(ws + OFF_DEC_WHH);
    unsigned short* outW_t   = (unsigned short*)(ws + OFF_OUTW);
    unsigned short* xbuf     = (unsigned short*)(ws + OFF_X);
    unsigned short* hbuf     = (unsigned short*)(ws + OFF_H);
    float*          cbuf     = (float*)(ws + OFF_C);
    float*          gbuf     = (float*)(ws + OFF_G);
    float*          logits   = (float*)(ws + OFF_LOGITS);
    int*            tok      = (int*)  (ws + OFF_TOK);

    // One-time (per call) bf16 swizzle of all weights.
    convert_tiled_bf16<<<32768, 256, 0, stream>>>(enc_Wih, encWih_t, 8192);
    convert_tiled_bf16<<<32768, 256, 0, stream>>>(enc_Whh, encWhh_t, 8192);
    convert_tiled_bf16<<<32768, 256, 0, stream>>>(dec_Wih, decWih_t, 8192);
    convert_tiled_bf16<<<32768, 256, 0, stream>>>(dec_Whh, decWhh_t, 8192);
    convert_tiled_bf16<<<128000, 256, 0, stream>>>(out_W,  outW_t,  32000);

    init_state<<<8, 256, 0, stream>>>(hbuf, cbuf, tok);

    // -------- encoder: 64 serial steps --------
    for (int t = 0; t < SEQ; ++t) {
        embed_bf16<<<4, 256, 0, stream>>>(emb_enc, src + t, xbuf);
        for (int l = 0; l < NL; ++l) {
            const unsigned short* inp = (l == 0) ? xbuf : hbuf;  // h[0] just updated
            wmma_gemv<<<64, 128, 0, stream>>>(
                encWih_t + (size_t)l * LAYER_WT_ELEMS, inp,
                encWhh_t + (size_t)l * LAYER_WT_ELEMS, hbuf + l * HID,
                enc_bih + l * GATES, enc_bhh + l * GATES, gbuf, 256);
            lstm_cell<<<4, 256, 0, stream>>>(gbuf, cbuf + l * HID, hbuf + l * HID);
        }
    }

    // -------- decoder: 64 serial greedy steps --------
    for (int t = 0; t < SEQ; ++t) {
        embed_bf16<<<4, 256, 0, stream>>>(emb_dec, tok, xbuf);
        for (int l = 0; l < NL; ++l) {
            const unsigned short* inp = (l == 0) ? xbuf : hbuf;
            wmma_gemv<<<64, 128, 0, stream>>>(
                decWih_t + (size_t)l * LAYER_WT_ELEMS, inp,
                decWhh_t + (size_t)l * LAYER_WT_ELEMS, hbuf + l * HID,
                dec_bih + l * GATES, dec_bhh + l * GATES, gbuf, 256);
            lstm_cell<<<4, 256, 0, stream>>>(gbuf, cbuf + l * HID, hbuf + l * HID);
        }
        // logits = out_W @ h[1] + out_b  (2000 row-blocks, exact waves)
        wmma_gemv<<<500, 128, 0, stream>>>(
            outW_t, hbuf + HID, (const unsigned short*)nullptr,
            (const unsigned short*)nullptr, out_b, (const float*)nullptr,
            logits, 2000);
        logsoftmax_argmax<<<1, 256, 0, stream>>>(logits, out + (size_t)t * VOCAB, tok);
    }
}
